// GATConv_61143154426041
// MI455X (gfx1250) — matrix operand order repaired and checked by reading, not compiled, not run on hardware
//
#include <hip/hip_runtime.h>
#include <hip/hip_bf16.h>

#define F_IN   512
#define F_OUT  256
#define ALPHA  0.2f

typedef __attribute__((ext_vector_type(16))) __bf16 v16bf;
typedef __attribute__((ext_vector_type(8)))  float  v8f;

// ---------- helpers ----------
// hardware packed fp32->bf16 (RNE) : one VALU op for two elements
__device__ __forceinline__ unsigned pack2bf(float lo, float hi) {
    unsigned r;
    asm("v_cvt_pk_bf16_f32 %0, %1, %2" : "=v"(r) : "v"(lo), "v"(hi));
    return r;
}

// order-preserving float<->uint key for atomic max over signed floats
__device__ __forceinline__ unsigned fkey(float f) {
    unsigned b = __float_as_uint(f);
    return (b & 0x80000000u) ? ~b : (b | 0x80000000u);
}
__device__ __forceinline__ float keyf(unsigned k) {
    unsigned b = (k & 0x80000000u) ? (k ^ 0x80000000u) : ~k;
    return __uint_as_float(b);
}

// two contiguous 16B LDS reads -> one 16-element bf16 fragment
__device__ __forceinline__ v16bf ld_frag(const unsigned short* p0,
                                         const unsigned short* p1) {
    union { uint4 q[2]; v16bf v; } u;
    u.q[0] = *(const uint4*)p0;
    u.q[1] = *(const uint4*)p1;
    return u.v;
}

// ---------- 0: zero per-row stats ----------
__global__ void gat_init(unsigned* rowmax, float* rowsum, unsigned* rowcnt,
                         float* colmean, int N) {
    int i = blockIdx.x * blockDim.x + threadIdx.x;
    if (i < N) { rowmax[i] = 0u; rowsum[i] = 0.0f; rowcnt[i] = 0u; }
    if (i < F_OUT) colmean[i] = 0.0f;
}

// ---------- 1: H = X @ W via bf16 WMMA, fp32 accumulate ----------
// grid (M/64, F_OUT/32), 256 threads (8 waves); each wave owns one 16x16 C tile.
// K staged 64-wide per barrier interval -> 2 WMMAs per interval,
// fragments assembled with ds_load_b128 (A row-major, B staged transposed).
#define KSTEP 64
#define ASTRIDE 72   // 144 bytes, multiple of 16
__global__ void gat_gemm_wmma(const float* __restrict__ X,
                              const float* __restrict__ W,
                              float* __restrict__ H) {
    __shared__ __align__(16) unsigned short As[64][ASTRIDE];   // 64 rows x 64 k (bf16)
    __shared__ __align__(16) unsigned short Bst[32][ASTRIDE];  // 32 cols x 64 k (transposed)

    const int tid     = threadIdx.x;
    const int lane    = tid & 31;
    const int wv      = tid >> 5;                 // 0..7
    const int rowBase = blockIdx.x * 64;
    const int colBase = blockIdx.y * 32;
    const int tileM   = (wv & 3) * 16;
    const int tileN   = (wv >> 2) * 16;

    v8f acc = {};

    const int mrow  = tileM + (lane & 15);
    const int ncol  = tileN + (lane & 15);
    const int koffA = (lane >> 4) << 3;           // 0 or 8
    const int koffB = (lane >> 4) << 4;           // 0 or 16

    for (int k0 = 0; k0 < F_IN; k0 += KSTEP) {
        // stage A: 64 rows x 64 k, float4 global loads -> packed bf16 b64 LDS stores
        #pragma unroll
        for (int i = 0; i < 4; ++i) {
            int base = (i * 256 + tid) * 4;
            int r = base >> 6, c = base & 63;
            float4 v = *(const float4*)&X[(size_t)(rowBase + r) * F_IN + k0 + c];
            uint2 p = { pack2bf(v.x, v.y), pack2bf(v.z, v.w) };
            *(uint2*)&As[r][c] = p;
        }
        // stage B: 64 k x 32 cols, float4 loads, store TRANSPOSED (col-major k-runs)
        #pragma unroll
        for (int i = 0; i < 2; ++i) {
            int base = (i * 256 + tid) * 4;
            int r = base >> 5, c = base & 31;     // r = k index, c = column
            float4 v = *(const float4*)&W[(size_t)(k0 + r) * F_OUT + colBase + c];
            unsigned p01 = pack2bf(v.x, v.y);
            unsigned p23 = pack2bf(v.z, v.w);
            Bst[c + 0][r] = (unsigned short)(p01 & 0xFFFFu);
            Bst[c + 1][r] = (unsigned short)(p01 >> 16);
            Bst[c + 2][r] = (unsigned short)(p23 & 0xFFFFu);
            Bst[c + 3][r] = (unsigned short)(p23 >> 16);
        }
        // prefetch next A K-tile into cache (gfx1250 global_prefetch_b8)
        if (k0 + KSTEP < F_IN)
            __builtin_prefetch(&X[(size_t)(rowBase + (tid >> 2)) * F_IN + k0 + KSTEP + (tid & 3) * 16], 0, 0);
        __syncthreads();

        #pragma unroll
        for (int kk = 0; kk < KSTEP; kk += 32) {
            // A frag: lane l<16 -> M=l, K={kk+koffA..+7, kk+16+koffA..+7}
            v16bf af = ld_frag(&As[mrow][kk + koffA], &As[mrow][kk + 16 + koffA]);
            // B frag: lane l -> col n=l&15, K={kk+koffA..+7, kk+16+koffA..+7}
            v16bf bf = ld_frag(&Bst[ncol][kk + koffA], &Bst[ncol][kk + 16 + koffA]);
            acc = __builtin_amdgcn_wmma_f32_16x16x32_bf16(
                      false, af, false, bf, (short)0, acc, false, false);
        }
        __syncthreads();
    }

    // store C: VGPR r of lane l -> M = tileM + (l>=16?8:0)+r, N = tileN + (l&15)
    const int col   = colBase + tileN + (lane & 15);
    const int rbase = rowBase + tileM + ((lane >> 4) << 3);
    #pragma unroll
    for (int r = 0; r < 8; ++r)
        H[(size_t)(rbase + r) * F_OUT + col] = acc[r];
}

// ---------- 2: s1[i]=h[i]·a[:256], s2[i]=h[i]·a[256:]  (one wave per node) ----------
__global__ void gat_node_scores(const float* __restrict__ H,
                                const float* __restrict__ a,
                                float* __restrict__ s1, float* __restrict__ s2, int N) {
    const int lane = threadIdx.x & 31;
    const int node = blockIdx.x * 8 + (threadIdx.x >> 5);
    if (node >= N) return;
    const float* hr = H + (size_t)node * F_OUT;
    float d1 = 0.f, d2 = 0.f;
    #pragma unroll
    for (int j = 0; j < F_OUT; j += 32) {
        float hv = hr[j + lane];
        d1 += hv * a[j + lane];
        d2 += hv * a[F_OUT + j + lane];
    }
    #pragma unroll
    for (int off = 16; off > 0; off >>= 1) {
        d1 += __shfl_xor(d1, off, 32);
        d2 += __shfl_xor(d2, off, 32);
    }
    if (lane == 0) { s1[node] = d1; s2[node] = d2; }
}

// ---------- 3: column mean of H (for rows with no edges: softmax -> uniform) ----------
__global__ void gat_colmean(const float* __restrict__ H, float* __restrict__ colmean, int N) {
    const int f  = threadIdx.x;                   // 0..255
    const int r0 = blockIdx.x * 32;
    float s = 0.f;
    #pragma unroll
    for (int i = 0; i < 32; ++i)
        s += H[(size_t)(r0 + i) * F_OUT + f];
    unsafeAtomicAdd(&colmean[f], s * (1.0f / (float)N));
}

// ---------- 4: per-edge score + row max + row count ----------
__global__ void gat_edge_score(const long long* __restrict__ edges,
                               const float* __restrict__ s1, const float* __restrict__ s2,
                               float* __restrict__ ew, unsigned* __restrict__ rowmax,
                               unsigned* __restrict__ rowcnt, long long E) {
    long long i = (long long)blockIdx.x * blockDim.x + threadIdx.x;
    if (i >= E) return;
    int src = (int)edges[i];
    int tgt = (int)edges[E + i];
    float v = s1[src] + s2[tgt];
    float e = v > 0.f ? v : ALPHA * v;            // leaky_relu
    ew[i] = e;
    atomicMax(&rowmax[src], fkey(e));
    atomicAdd(&rowcnt[src], 1u);
}

// ---------- 5: per-edge exp(e - rowmax) + row sum ----------
__global__ void gat_edge_exp(const long long* __restrict__ edges,
                             float* __restrict__ ew, const unsigned* __restrict__ rowmax,
                             float* __restrict__ rowsum, long long E) {
    long long i = (long long)blockIdx.x * blockDim.x + threadIdx.x;
    if (i >= E) return;
    int src = (int)edges[i];
    float w = __expf(ew[i] - keyf(rowmax[src]));
    ew[i] = w;
    unsafeAtomicAdd(&rowsum[src], w);
}

// ---------- 6: out = bias (broadcast init) ----------
__global__ void gat_out_init(float* __restrict__ out, const float* __restrict__ bias, int N) {
    size_t idx = (size_t)blockIdx.x * blockDim.x + threadIdx.x;
    if (idx < (size_t)N * F_OUT) out[idx] = bias[idx & (F_OUT - 1)];
}

// ---------- 7: sparse aggregate: out[src] += (w/rowsum[src]) * h[tgt]  (one wave per edge) ----------
__global__ void gat_scatter(const long long* __restrict__ edges,
                            const float* __restrict__ ew, const float* __restrict__ rowsum,
                            const float* __restrict__ H, float* __restrict__ out, long long E) {
    const int  lane = threadIdx.x & 31;
    long long  i    = (long long)blockIdx.x * 8 + (threadIdx.x >> 5);
    if (i >= E) return;
    int src = (int)edges[i];
    int tgt = (int)edges[E + i];
    float coef = ew[i] / rowsum[src];
    const float* hr = H + (size_t)tgt * F_OUT;
    float* orow = out + (size_t)src * F_OUT;
    #pragma unroll
    for (int j = 0; j < F_OUT; j += 32)
        unsafeAtomicAdd(&orow[j + lane], coef * hr[j + lane]);
}

// ---------- 8: rows with no outgoing edges -> uniform softmax = colmean + bias ----------
__global__ void gat_fixup(const unsigned* __restrict__ rowcnt,
                          const float* __restrict__ colmean, const float* __restrict__ bias,
                          float* __restrict__ out, int N) {
    int row = blockIdx.x;
    if (row >= N || rowcnt[row] != 0u) return;
    int f = threadIdx.x;
    out[(size_t)row * F_OUT + f] = colmean[f] + bias[f];
}

extern "C" void kernel_launch(void* const* d_in, const int* in_sizes, int n_in,
                              void* d_out, int out_size, void* d_ws, size_t ws_size,
                              hipStream_t stream) {
    const float*     x      = (const float*)d_in[0];
    const long long* edges  = (const long long*)d_in[1];   // int64 [2, E]
    const float*     weight = (const float*)d_in[2];
    const float*     a      = (const float*)d_in[3];
    const float*     bias   = (const float*)d_in[4];
    float*           out    = (float*)d_out;

    const int       N = in_sizes[0] / F_IN;    // 8192
    const long long E = in_sizes[1] / 2;       // 262144

    // workspace layout (fp32 words)
    float*    h       = (float*)d_ws;                          // N*F_OUT
    float*    ew      = h + (size_t)N * F_OUT;                 // E
    float*    s1      = ew + E;                                // N
    float*    s2      = s1 + N;                                // N
    float*    rowsum  = s2 + N;                                // N
    float*    colmean = rowsum + N;                            // F_OUT
    unsigned* rowmax  = (unsigned*)(colmean + F_OUT);          // N
    unsigned* rowcnt  = rowmax + N;                            // N

    gat_init<<<(N + 255) / 256, 256, 0, stream>>>(rowmax, rowsum, rowcnt, colmean, N);

    dim3 ggrid(N / 64, F_OUT / 32);
    gat_gemm_wmma<<<ggrid, 256, 0, stream>>>(x, weight, h);

    gat_node_scores<<<(N + 7) / 8, 256, 0, stream>>>(h, a, s1, s2, N);
    gat_colmean<<<N / 32, F_OUT, 0, stream>>>(h, colmean, N);

    int eblocks = (int)((E + 255) / 256);
    gat_edge_score<<<eblocks, 256, 0, stream>>>(edges, s1, s2, ew, rowmax, rowcnt, E);
    gat_edge_exp<<<eblocks, 256, 0, stream>>>(edges, ew, rowmax, rowsum, E);

    gat_out_init<<<(int)(((size_t)N * F_OUT + 255) / 256), 256, 0, stream>>>(out, bias, N);

    int sblocks = (int)((E + 7) / 8);
    gat_scatter<<<sblocks, 256, 0, stream>>>(edges, ew, rowsum, h, out, E);

    gat_fixup<<<N, F_OUT, 0, stream>>>(rowcnt, colmean, bias, out, N);
}
